// Decoder_14199161881169
// MI455X (gfx1250) — compile-verified, matrix operand order checked by reading, not run on hardware
//
#include <hip/hip_runtime.h>
#include <hip/hip_bf16.h>

#define Bsz  128
#define TEs  256
#define Ts   200
#define Hs   256
#define PREs 128
#define OUTs 400
#define KBIG (1 << 30)

typedef __attribute__((ext_vector_type(16))) __bf16          v16bf;
typedef __attribute__((ext_vector_type(16))) unsigned short  v16u;
typedef __attribute__((ext_vector_type(8)))  float           v8f;

union BF16x16 { v16u u; v16bf b; };

__device__ __forceinline__ unsigned short f2bf(float f) {
  unsigned int x = __float_as_uint(f);
  x += 0x7FFFu + ((x >> 16) & 1u);
  return (unsigned short)(x >> 16);
}

__device__ __forceinline__ float fast_sigm(float x) {
  return 1.f / (1.f + __expf(-x));
}
__device__ __forceinline__ float fast_tanh(float x) {
  float e = __expf(2.f * x);
  return 1.f - 2.f / (e + 1.f);
}

// ---------------------------------------------------------------------------
// A-fragment: 16x32 bf16 tile (M x K) from f32 rows. Unguarded straight-line
// loads; conversion via native f32->bf16 casts (packed cvt on gfx1250).
// Lane L / L+16 hold row M = L&15.
// ---------------------------------------------------------------------------
__device__ __forceinline__ v16bf afrag_f32(const float* X, long ld,
                                           int kbase, int lane) {
  BF16x16 t;
  const float* xr = X + (long)(lane & 15) * ld + kbase + ((lane >> 4) << 3);
#pragma unroll
  for (int i = 0; i < 8; ++i) {
    int k0 = ((i < 4) ? 0 : 16) + ((i & 3) << 1);
    t.b[2 * i]     = (__bf16)xr[k0];
    t.b[2 * i + 1] = (__bf16)xr[k0 + 1];
  }
  return t.b;
}

// Branchless K-clamped variant (only for the zero-padded tail tile):
// unconditional loads at clamped addresses + cndmask zeroing, no branches.
__device__ __forceinline__ v16bf afrag_f32_clamp(const float* X, long ld,
                                                 int kbase, int kmax, int lane) {
  BF16x16 t;
  int row = lane & 15;
  int kh  = (lane >> 4) << 3;
  const float* xr = X + (long)row * ld;
#pragma unroll
  for (int i = 0; i < 8; ++i) {
    int k0 = kbase + ((i < 4) ? 0 : 16) + kh + ((i & 3) << 1);
    int k1 = k0 + 1;
    float f0 = xr[k0 < kmax ? k0 : 0];
    float f1 = xr[k1 < kmax ? k1 : 0];
    f0 = (k0 < kmax) ? f0 : 0.f;
    f1 = (k1 < kmax) ? f1 : 0.f;
    t.b[2 * i]     = (__bf16)f0;
    t.b[2 * i + 1] = (__bf16)f1;
  }
  return t.b;
}

// B-fragment: pre-packed bf16 weights; tile = 32 lanes x 16 bf16 contiguous.
__device__ __forceinline__ v16bf load_bfrag(const unsigned short* Wpk,
                                            long tileIdx, int lane) {
  BF16x16 t;
  t.u = *(const v16u*)(Wpk + (tileIdx * 32 + lane) * 16);
  return t.b;
}

// C-tile store: D layout lane -> (col = lane&15, rows mb..mb+7).
template <bool BIAS, bool RELU>
__device__ __forceinline__ void store_ctile(const v8f& acc, int nt, int lane,
                                            const float* bias,
                                            float* Y, long ldy) {
  int   col = nt * 16 + (lane & 15);
  int   mb  = (lane >> 4) << 3;
  float bv  = BIAS ? bias[col] : 0.f;
#pragma unroll
  for (int j = 0; j < 8; ++j) {
    float v = acc[j];
    if (BIAS) v += bv;
    if (RELU) v = fmaxf(v, 0.f);
    Y[(long)(mb + j) * ldy + col] = v;
  }
}

// ---------------------------------------------------------------------------
// General GEMM: Y[16, Ntiles*16] = act( concat(Xa,Xb)[16,K] @ Wpk + bias ).
// Wave id is scalarized (readfirstlane) so tile guards are pure scalar
// branches and EXEC is provably all-1s at every WMMA. Each wave processes two
// N-tiles per pass so every A fragment feeds 2 WMMAs.
// ---------------------------------------------------------------------------
template <bool BIAS, bool RELU>
__device__ void wmma_gemm(const float* Xa, long lda, int KaT, int kmaxA,
                          const float* Xb, long ldb, int KbT,
                          const unsigned short* Wpk, int Ntiles,
                          const float* bias,
                          float* Y, long ldy) {
  int lane  = threadIdx.x & 31;
  int swave = __builtin_amdgcn_readfirstlane(threadIdx.x) >> 5;
  int nW    = blockDim.x >> 5;
  int KT    = KaT + KbT;
  for (int nt0 = swave; nt0 < Ntiles; nt0 += 2 * nW) {
    int  nt1 = nt0 + nW;
    bool v1  = nt1 < Ntiles;
    v8f acc0 = {}, acc1 = {};
    for (int kt = 0; kt < KT; ++kt) {
      v16bf a;
      if (kt < KaT) {
        int kb = kt << 5;
        a = (kb + 32 <= kmaxA) ? afrag_f32(Xa, lda, kb, lane)
                               : afrag_f32_clamp(Xa, lda, kb, kmaxA, lane);
      } else {
        a = afrag_f32(Xb, ldb, (kt - KaT) << 5, lane);
      }
      acc0 = __builtin_amdgcn_wmma_f32_16x16x32_bf16(false, a, false,
               load_bfrag(Wpk, (long)nt0 * KT + kt, lane), (short)0, acc0, false, false);
      if (v1)
        acc1 = __builtin_amdgcn_wmma_f32_16x16x32_bf16(false, a, false,
                 load_bfrag(Wpk, (long)nt1 * KT + kt, lane), (short)0, acc1, false, false);
    }
    store_ctile<BIAS, RELU>(acc0, nt0, lane, bias, Y, ldy);
    if (v1) store_ctile<BIAS, RELU>(acc1, nt1, lane, bias, Y, ldy);
  }
}

// GRU gate combine for one N-tile (in registers).
__device__ __forceinline__ void gru_combine(int nt, int lane,
    const v8f& ir, const v8f& iz, const v8f& in,
    const v8f& hr, const v8f& hz, const v8f& hn,
    const float* bih, const float* bhh,
    const float* hOld, float* hNew) {
  int col = nt * 16 + (lane & 15);
  int mb  = (lane >> 4) << 3;
  float bir = bih[col],       bhr = bhh[col];
  float biz = bih[256 + col], bhz = bhh[256 + col];
  float bin = bih[512 + col], bhn = bhh[512 + col];
#pragma unroll
  for (int j = 0; j < 8; ++j) {
    int   m  = mb + j;
    float r  = fast_sigm(ir[j] + bir + hr[j] + bhr);
    float z  = fast_sigm(iz[j] + biz + hz[j] + bhz);
    float n  = fast_tanh(in[j] + bin + r * (hn[j] + bhn));
    float ho = hOld[m * 256 + col];
    hNew[m * 256 + col] = (1.f - z) * n + z * ho;
  }
}

// ---------------------------------------------------------------------------
// Fused GRU cell: hNew[16,256] from x=concat(Xa,Xb) and hOld[16,256].
// 12 gate accumulators (2 N-tiles x r/z/n x ih/hh) live in registers; each
// A fragment feeds 6 WMMAs. No [16,768] gate buffers in LDS.
// ---------------------------------------------------------------------------
__device__ void wmma_gru(const float* Xa, long lda, int KaT,
                         const float* Xb, long ldb, int KbT,
                         const unsigned short* Wih, const unsigned short* Whh,
                         const float* bih, const float* bhh,
                         const float* hOld, float* hNew) {
  int lane  = threadIdx.x & 31;
  int swave = __builtin_amdgcn_readfirstlane(threadIdx.x) >> 5;
  int nW    = blockDim.x >> 5;
  int KT    = KaT + KbT;
  for (int nt0 = swave; nt0 < 16; nt0 += 2 * nW) {
    int  nt1 = nt0 + nW;
    bool v1  = nt1 < 16;
    v8f ir0 = {}, iz0 = {}, in0 = {}, hr0 = {}, hz0 = {}, hn0 = {};
    v8f ir1 = {}, iz1 = {}, in1 = {}, hr1 = {}, hz1 = {}, hn1 = {};
    for (int kt = 0; kt < KT; ++kt) {
      v16bf a = (kt < KaT) ? afrag_f32(Xa, lda, kt << 5, lane)
                           : afrag_f32(Xb, ldb, (kt - KaT) << 5, lane);
      ir0 = __builtin_amdgcn_wmma_f32_16x16x32_bf16(false, a, false,
              load_bfrag(Wih, (long)(nt0)      * KT + kt, lane), (short)0, ir0, false, false);
      iz0 = __builtin_amdgcn_wmma_f32_16x16x32_bf16(false, a, false,
              load_bfrag(Wih, (long)(16 + nt0) * KT + kt, lane), (short)0, iz0, false, false);
      in0 = __builtin_amdgcn_wmma_f32_16x16x32_bf16(false, a, false,
              load_bfrag(Wih, (long)(32 + nt0) * KT + kt, lane), (short)0, in0, false, false);
      if (v1) {
        ir1 = __builtin_amdgcn_wmma_f32_16x16x32_bf16(false, a, false,
                load_bfrag(Wih, (long)(nt1)      * KT + kt, lane), (short)0, ir1, false, false);
        iz1 = __builtin_amdgcn_wmma_f32_16x16x32_bf16(false, a, false,
                load_bfrag(Wih, (long)(16 + nt1) * KT + kt, lane), (short)0, iz1, false, false);
        in1 = __builtin_amdgcn_wmma_f32_16x16x32_bf16(false, a, false,
                load_bfrag(Wih, (long)(32 + nt1) * KT + kt, lane), (short)0, in1, false, false);
      }
    }
    for (int kt = 0; kt < 8; ++kt) {
      v16bf a = afrag_f32(hOld, 256, kt << 5, lane);
      hr0 = __builtin_amdgcn_wmma_f32_16x16x32_bf16(false, a, false,
              load_bfrag(Whh, (long)(nt0)      * 8 + kt, lane), (short)0, hr0, false, false);
      hz0 = __builtin_amdgcn_wmma_f32_16x16x32_bf16(false, a, false,
              load_bfrag(Whh, (long)(16 + nt0) * 8 + kt, lane), (short)0, hz0, false, false);
      hn0 = __builtin_amdgcn_wmma_f32_16x16x32_bf16(false, a, false,
              load_bfrag(Whh, (long)(32 + nt0) * 8 + kt, lane), (short)0, hn0, false, false);
      if (v1) {
        hr1 = __builtin_amdgcn_wmma_f32_16x16x32_bf16(false, a, false,
                load_bfrag(Whh, (long)(nt1)      * 8 + kt, lane), (short)0, hr1, false, false);
        hz1 = __builtin_amdgcn_wmma_f32_16x16x32_bf16(false, a, false,
                load_bfrag(Whh, (long)(16 + nt1) * 8 + kt, lane), (short)0, hz1, false, false);
        hn1 = __builtin_amdgcn_wmma_f32_16x16x32_bf16(false, a, false,
                load_bfrag(Whh, (long)(32 + nt1) * 8 + kt, lane), (short)0, hn1, false, false);
      }
    }
    gru_combine(nt0, lane, ir0, iz0, in0, hr0, hz0, hn0, bih, bhh, hOld, hNew);
    if (v1) gru_combine(nt1, lane, ir1, iz1, in1, hr1, hz1, hn1, bih, bhh, hOld, hNew);
  }
}

// ---------------------------------------------------------------------------
// Pack f32 weight [N,K] (row-major, as used in x @ W.T) into fragment-ready
// bf16 tiles: [ntile][ktile][lane(32)][16], K zero-padded to Ktiles*32.
// ---------------------------------------------------------------------------
__global__ void pack_w(const float* __restrict__ src, unsigned short* __restrict__ dst,
                       int N, int K, int Ktiles) {
  int  Ntiles = N >> 4;
  long total  = (long)Ntiles * Ktiles * 512;
  for (long idx = (long)blockIdx.x * blockDim.x + threadIdx.x; idx < total;
       idx += (long)gridDim.x * blockDim.x) {
    int  j    = (int)(idx & 15);
    int  lane = (int)((idx >> 4) & 31);
    long tile = idx >> 9;
    int  kt   = (int)(tile % Ktiles);
    int  nt   = (int)(tile / Ktiles);
    int  n    = (nt << 4) + (lane & 15);
    int  i    = j >> 1, e = j & 1;
    int  k    = (kt << 5) + ((i < 4) ? 0 : 16) + ((lane >> 4) << 3) + ((i & 3) << 1) + e;
    float v   = (k < K) ? src[(long)n * K + k] : 0.f;
    dst[idx]  = f2bf(v);
  }
}

// pm = enc @ wm.T : [B*TE, 256] x [256,256] -> f32, one M-tile per block.
__global__ __launch_bounds__(256) void pm_kernel(const float* __restrict__ enc,
                                                 const unsigned short* __restrict__ wmp,
                                                 float* __restrict__ pm) {
  long r0 = (long)blockIdx.x * 16;
  wmma_gemm<false, false>(enc + r0 * 256, 256, 8, KBIG, nullptr, 0, 0,
                          wmp, 16, nullptr, pm + r0 * 256, 256);
}

// ---------------------------------------------------------------------------
// Persistent decoder: one block = 16 batch rows, runs all 200 steps.
// ---------------------------------------------------------------------------
__global__ __launch_bounds__(256) void decoder_kernel(
    const float* __restrict__ enc, const float* __restrict__ targets,
    const unsigned short* w1p, const float* b1,
    const unsigned short* w2p, const float* b2,
    const unsigned short* wihp, const unsigned short* whhp,
    const float* bih, const float* bhh,
    const unsigned short* wqp, const float* v_attn,
    const unsigned short* wpp, const float* bp,
    const unsigned short* g1w, const unsigned short* g1u,
    const float* g1bi, const float* g1bh,
    const unsigned short* g2w, const unsigned short* g2u,
    const float* g2bi, const float* g2bh,
    const unsigned short* wop, const float* bo,
    const float* __restrict__ pm,
    float* __restrict__ outputs, float* __restrict__ aligns) {
  __shared__ float ah[2][16 * 256];   // attention-RNN hidden (double buffered)
  __shared__ float h1[2][16 * 256];
  __shared__ float h2[2][16 * 256];
  __shared__ float av[16 * 256];      // attention context
  __shared__ float R1[16 * 256];      // P1 -> Q -> dec_in
  __shared__ float R2[16 * 256];      // P2 -> align -> dec2/dec3
  __shared__ float sv[256];           // v_attn

  int b0    = blockIdx.x * 16;
  int tid   = threadIdx.x;
  int lane  = tid & 31;
  int swave = __builtin_amdgcn_readfirstlane(tid) >> 5;

  for (int i = tid; i < 16 * 256; i += blockDim.x) {
    ah[0][i] = 0.f; h1[0][i] = 0.f; h2[0][i] = 0.f; av[i] = 0.f;
  }
  for (int i = tid; i < 256; i += blockDim.x) sv[i] = v_attn[i];
  __syncthreads();

  for (int t = 0; t < Ts; ++t) {
    int c = t & 1, n = c ^ 1;

    // S1: prenet layer 1. t==0 frame is all-zero -> p1 = relu(b1) directly.
    if (t == 0) {
      for (int i = tid; i < 16 * 256; i += blockDim.x)
        R1[i] = fmaxf(b1[i & 255], 0.f);
    } else {
      wmma_gemm<true, true>(targets + ((long)b0 * Ts + (t - 1)) * OUTs,
                            (long)Ts * OUTs, 13, OUTs, nullptr, 0, 0,
                            w1p, 16, b1, R1, 256);
    }
    __syncthreads();

    // S2: prenet layer 2 -> P2 [16,128] in R2
    wmma_gemm<true, true>(R1, 256, 8, KBIG, nullptr, 0, 0, w2p, 8, b2, R2, 128);
    __syncthreads();

    // S3: attention GRU, x = concat(P2, attn_val)
    wmma_gru(R2, 128, 4, av, 256, 8, wihp, whhp, bih, bhh, ah[c], ah[n]);
    __syncthreads();

    // S4: q = attn_h @ wq.T -> R1
    wmma_gemm<false, false>(ah[n], 256, 8, KBIG, nullptr, 0, 0, wqp, 16,
                            nullptr, R1, 256);
    __syncthreads();

    // S5: Bahdanau scores + softmax -> R2 (align) + global alignments
    for (int m = swave; m < 16; m += (blockDim.x >> 5)) {
      int   b = b0 + m;
      float ev[8];
      float mx = -1e30f;
#pragma unroll
      for (int it = 0; it < 8; ++it) {
        int tp = lane + 32 * it;
        const float* pmr = pm + ((long)b * TEs + tp) * Hs;
        float s = 0.f;
        for (int h = 0; h < Hs; ++h)
          s += sv[h] * fast_tanh(pmr[h] + R1[m * 256 + h]);
        ev[it] = s;
        mx = fmaxf(mx, s);
      }
      for (int off = 16; off; off >>= 1) mx = fmaxf(mx, __shfl_xor(mx, off, 32));
      float sum = 0.f;
#pragma unroll
      for (int it = 0; it < 8; ++it) { ev[it] = __expf(ev[it] - mx); sum += ev[it]; }
      for (int off = 16; off; off >>= 1) sum += __shfl_xor(sum, off, 32);
      float inv = 1.f / sum;
#pragma unroll
      for (int it = 0; it < 8; ++it) {
        int   tp = lane + 32 * it;
        float a  = ev[it] * inv;
        R2[m * 256 + tp] = a;
        aligns[((long)b * Ts + t) * TEs + tp] = a;
      }
    }
    __syncthreads();

    // S6: attn_val[m,h] = sum_t align[m,t] * enc[b0+m,t,h]  (per-row B matrix)
    {
      int m  = tid >> 4;
      int hb = (tid & 15) << 4;
      float acc[16];
#pragma unroll
      for (int j = 0; j < 16; ++j) acc[j] = 0.f;
      const float* encb = enc + ((long)(b0 + m) * TEs) * Hs;
      for (int tt = 0; tt < TEs; ++tt) {
        float a = R2[m * 256 + tt];
        const float* er = encb + (long)tt * Hs + hb;
#pragma unroll
        for (int j = 0; j < 16; ++j) acc[j] += a * er[j];
      }
#pragma unroll
      for (int j = 0; j < 16; ++j) av[m * 256 + hb + j] = acc[j];
    }
    __syncthreads();

    // S7: dec_in = concat(attn_h, attn_val) @ wp.T + bp -> R1
    wmma_gemm<true, false>(ah[n], 256, 8, KBIG, av, 256, 8, wpp, 16, bp, R1, 256);
    __syncthreads();

    // S8: decoder GRU 1
    wmma_gru(R1, 256, 8, nullptr, 0, 0, g1w, g1u, g1bi, g1bh, h1[c], h1[n]);
    __syncthreads();

    // S9: dec2 = h1 + dec_in -> R2
    for (int i = tid; i < 16 * 256; i += blockDim.x) R2[i] = h1[n][i] + R1[i];
    __syncthreads();

    // S10: decoder GRU 2
    wmma_gru(R2, 256, 8, nullptr, 0, 0, g2w, g2u, g2bi, g2bh, h2[c], h2[n]);
    __syncthreads();

    // S11: dec3 = h2 + dec2 (in place)
    for (int i = tid; i < 16 * 256; i += blockDim.x) R2[i] += h2[n][i];
    __syncthreads();

    // S12: out = dec3 @ wo.T + bo -> global outputs [B,T,400]
    wmma_gemm<true, false>(R2, 256, 8, KBIG, nullptr, 0, 0, wop, 25, bo,
                           outputs + ((long)b0 * Ts + t) * OUTs, (long)Ts * OUTs);
    __syncthreads();
  }
}

// ---------------------------------------------------------------------------
extern "C" void kernel_launch(void* const* d_in, const int* in_sizes, int n_in,
                              void* d_out, int out_size, void* d_ws, size_t ws_size,
                              hipStream_t stream) {
  const float* enc     = (const float*)d_in[0];
  const float* targets = (const float*)d_in[1];

  // workspace carve: pm (f32) then packed bf16 weights
  float* pm = (float*)d_ws;
  const long PM_ELEMS = (long)Bsz * TEs * Hs;          // 8,388,608 f32
  unsigned short* wpk = (unsigned short*)(pm + PM_ELEMS);

  unsigned short* w1p = wpk;                  // 256x416 -> 16*13*512
  unsigned short* w2p = w1p + 106496;         // 128x256 ->  8* 8*512
  unsigned short* wih = w2p + 32768;          // 768x384 -> 48*12*512
  unsigned short* whh = wih + 294912;         // 768x256 -> 48* 8*512
  unsigned short* wqp = whh + 196608;         // 256x256
  unsigned short* wmp = wqp + 65536;          // 256x256
  unsigned short* wpp = wmp + 65536;          // 256x512
  unsigned short* g1w = wpp + 131072;         // 768x256
  unsigned short* g1u = g1w + 196608;
  unsigned short* g2w = g1u + 196608;
  unsigned short* g2u = g2w + 196608;
  unsigned short* wop = g2u + 196608;         // 400x256 -> 25*8*512

  pack_w<<<512, 256, 0, stream>>>((const float*)d_in[2],  w1p, 256, 400, 13);
  pack_w<<<512, 256, 0, stream>>>((const float*)d_in[4],  w2p, 128, 256, 8);
  pack_w<<<512, 256, 0, stream>>>((const float*)d_in[6],  wih, 768, 384, 12);
  pack_w<<<512, 256, 0, stream>>>((const float*)d_in[7],  whh, 768, 256, 8);
  pack_w<<<512, 256, 0, stream>>>((const float*)d_in[10], wqp, 256, 256, 8);
  pack_w<<<512, 256, 0, stream>>>((const float*)d_in[11], wmp, 256, 256, 8);
  pack_w<<<512, 256, 0, stream>>>((const float*)d_in[13], wpp, 256, 512, 16);
  pack_w<<<512, 256, 0, stream>>>((const float*)d_in[15], g1w, 768, 256, 8);
  pack_w<<<512, 256, 0, stream>>>((const float*)d_in[16], g1u, 768, 256, 8);
  pack_w<<<512, 256, 0, stream>>>((const float*)d_in[19], g2w, 768, 256, 8);
  pack_w<<<512, 256, 0, stream>>>((const float*)d_in[20], g2u, 768, 256, 8);
  pack_w<<<512, 256, 0, stream>>>((const float*)d_in[23], wop, 400, 256, 8);

  // pm = enc @ wm.T  (B*TE / 16 = 2048 M-tiles)
  pm_kernel<<<(Bsz * TEs) / 16, 256, 0, stream>>>(enc, wmp, pm);

  float* outputs = (float*)d_out;
  float* aligns  = outputs + (long)Bsz * Ts * OUTs;

  decoder_kernel<<<Bsz / 16, 256, 0, stream>>>(
      enc, targets,
      w1p, (const float*)d_in[3],
      w2p, (const float*)d_in[5],
      wih, whh, (const float*)d_in[8], (const float*)d_in[9],
      wqp, (const float*)d_in[12],
      wpp, (const float*)d_in[14],
      g1w, g1u, (const float*)d_in[17], (const float*)d_in[18],
      g2w, g2u, (const float*)d_in[21], (const float*)d_in[22],
      wop, (const float*)d_in[24],
      pm, outputs, aligns);
}
